// EnhancedMPGNN_26869315404034
// MI455X (gfx1250) — matrix operand order repaired and checked by reading, not compiled, not run on hardware
//
#include <hip/hip_runtime.h>
#include <hip/hip_bf16.h>
#include <float.h>

#define N_USER_C 50000
#define N_MOVIE_C 50000
#define NN 100000
#define NE 500000
#define ET 600000        // NE + NN self loops
#define HDIM 128
#define LN_EPS 1e-5f
#define COS_EPS 1e-8f

typedef _Float16 half_t;
typedef __attribute__((ext_vector_type(16))) _Float16 v16h;
typedef __attribute__((ext_vector_type(8)))  float    v8f;

// ---------------- WMMA fragment helpers (16x16x32 f16, f32 accum) -----------
// A: 16x32 tile, row-major, leading dim ld (halves).
// Lane L<16 : M=L,    K = kbase+0..15
// Lane L>=16: M=L-16, K = kbase+16..31
__device__ __forceinline__ v16h load_a16(const half_t* A, int ld, int kbase) {
  int lane = threadIdx.x & 31;
  const half_t* p = A + (size_t)(lane & 15) * ld + kbase + ((lane >> 4) << 4);
  v16h a;
#pragma unroll
  for (int j = 0; j < 16; ++j) a[j] = p[j];
  return a;
}
// B: 32x16 tile taken from W (row-major, (out,in), leading dim ld).
// Lane L<16 : N=L,    K = kbase+0..15 ; Lane L>=16: N=L-16, K = kbase+16..31
__device__ __forceinline__ v16h load_b16(const half_t* W, int ld, int nbase, int kbase) {
  int lane = threadIdx.x & 31;
  const half_t* p = W + (size_t)(nbase + (lane & 15)) * ld + kbase + ((lane >> 4) << 4);
  v16h b;
#pragma unroll
  for (int j = 0; j < 16; ++j) b[j] = p[j];
  return b;
}
__device__ __forceinline__ v8f wmma32(v16h a, v16h b, v8f c) {
  return __builtin_amdgcn_wmma_f32_16x16x32_f16(false, a, false, b, (short)0, c, false, false);
}
// C/D: element r of v8f at lane l -> M = r + 8*(l>=16), N = l&15

// ---------------- misc helpers ----------------------------------------------
__device__ __forceinline__ unsigned f2ord(float f) {
  unsigned u = __float_as_uint(f);
  return (u & 0x80000000u) ? ~u : (u | 0x80000000u);
}
__device__ __forceinline__ float ord2f(unsigned u) {
  return (u & 0x80000000u) ? __uint_as_float(u & 0x7FFFFFFFu) : __uint_as_float(~u);
}

// ---------------- kernels ---------------------------------------------------
__global__ void k_f32_to_f16(const float* __restrict__ s, half_t* __restrict__ d, int n) {
  int i = blockIdx.x * blockDim.x + threadIdx.x;
  if (i < n) d[i] = (half_t)s[i];
}

__global__ void k_zero(float* __restrict__ p, int n) {
  int i = blockIdx.x * blockDim.x + threadIdx.x;
  if (i < n) p[i] = 0.f;
}

// encoder: relu(x@W1.T+b1) @ W2.T + b2, then LayerNorm. 16 nodes / block, 128 thr.
__global__ __launch_bounds__(128)
void k_encoder(const float* __restrict__ xin, int fin, int nNodes, int nodeBase,
               const half_t* __restrict__ W1, const float* __restrict__ b1,
               const half_t* __restrict__ W2, const float* __restrict__ b2,
               const float* __restrict__ g, const float* __restrict__ beta,
               float* __restrict__ xout, half_t* __restrict__ xhout) {
  __shared__ half_t sA[16 * 64];
  __shared__ half_t sH[16 * HDIM];
  __shared__ float  sY[16 * HDIM];
  __shared__ float  sMu[16], sRs[16];

  int nb = blockIdx.x * 16;
  for (int idx = threadIdx.x; idx < 16 * fin; idx += 128) {
    int r = idx / fin, c = idx - r * fin;
    int node = nb + r;
    sA[r * fin + c] = (node < nNodes) ? (half_t)xin[(size_t)node * fin + c] : (half_t)0.f;
  }
  __syncthreads();

  int wave = threadIdx.x >> 5, lane = threadIdx.x & 31;
  // layer 1: fin -> 128, relu
  for (int t = wave * 2; t < wave * 2 + 2; ++t) {
    v8f acc = {};
    for (int k = 0; k < fin; k += 32)
      acc = wmma32(load_a16(sA, fin, k), load_b16(W1, fin, t * 16, k), acc);
    int n = t * 16 + (lane & 15), M0 = (lane >> 4) * 8;
    float bias = b1[n];
#pragma unroll
    for (int r = 0; r < 8; ++r)
      sH[(size_t)(M0 + r) * HDIM + n] = (half_t)fmaxf(acc[r] + bias, 0.f);
  }
  __syncthreads();
  // layer 2: 128 -> 128
  for (int t = wave * 2; t < wave * 2 + 2; ++t) {
    v8f acc = {};
#pragma unroll
    for (int k = 0; k < HDIM; k += 32)
      acc = wmma32(load_a16(sH, HDIM, k), load_b16(W2, HDIM, t * 16, k), acc);
    int n = t * 16 + (lane & 15), M0 = (lane >> 4) * 8;
    float bias = b2[n];
#pragma unroll
    for (int r = 0; r < 8; ++r)
      sY[(size_t)(M0 + r) * HDIM + n] = acc[r] + bias;
  }
  __syncthreads();
  // layer norm stats
  if (threadIdx.x < 16) {
    float s = 0.f;
    for (int c = 0; c < HDIM; ++c) s += sY[threadIdx.x * HDIM + c];
    float mu = s / HDIM, v = 0.f;
    for (int c = 0; c < HDIM; ++c) { float d = sY[threadIdx.x * HDIM + c] - mu; v += d * d; }
    sMu[threadIdx.x] = mu;
    sRs[threadIdx.x] = rsqrtf(v / HDIM + LN_EPS);
  }
  __syncthreads();
  for (int idx = threadIdx.x; idx < 16 * HDIM; idx += 128) {
    int r = idx >> 7, c = idx & (HDIM - 1);
    int node = nb + r;
    if (node < nNodes) {
      float v = (sY[idx] - sMu[r]) * sRs[r] * g[c] + beta[c];
      xout[(size_t)(nodeBase + node) * HDIM + c] = v;
      xhout[(size_t)(nodeBase + node) * HDIM + c] = (half_t)v;
    }
  }
}

// y = xh @ W.T + b over all NN nodes.
// 80 nodes (5 M-tiles) per block; each wave holds its 8 weight fragments in
// VGPRs for the whole block, so the K-loop is register-resident WMMA.
__global__ __launch_bounds__(128)
void k_linear(const half_t* __restrict__ xh, const half_t* __restrict__ W,
              const float* __restrict__ b, float* __restrict__ y) {
  int nb = blockIdx.x * 80;
  int wave = threadIdx.x >> 5, lane = threadIdx.x & 31;

  v16h B[2][4];
#pragma unroll
  for (int tt = 0; tt < 2; ++tt)
#pragma unroll
    for (int k = 0; k < 4; ++k)
      B[tt][k] = load_b16(W, HDIM, (wave * 2 + tt) * 16, k * 32);
  float bias[2];
  bias[0] = b[wave * 32 + (lane & 15)];
  bias[1] = b[wave * 32 + 16 + (lane & 15)];

  for (int m = 0; m < 5; ++m) {
    const half_t* A = xh + (size_t)(nb + m * 16) * HDIM;
    v16h a[4];
#pragma unroll
    for (int k = 0; k < 4; ++k) a[k] = load_a16(A, HDIM, k * 32);
#pragma unroll
    for (int tt = 0; tt < 2; ++tt) {
      v8f acc = {};
#pragma unroll
      for (int k = 0; k < 4; ++k) acc = wmma32(a[k], B[tt][k], acc);
      int n = (wave * 2 + tt) * 16 + (lane & 15), M0 = (lane >> 4) * 8;
#pragma unroll
      for (int r = 0; r < 8; ++r)
        y[(size_t)(nb + m * 16 + M0 + r) * HDIM + n] = acc[r] + bias[tt];
    }
  }
}

__global__ void k_norms(const float* __restrict__ x, float* __restrict__ norms) {
  int n = blockIdx.x * blockDim.x + threadIdx.x;
  if (n >= NN) return;
  const float4* p = (const float4*)(x + (size_t)n * HDIM);
  float s = 0.f;
#pragma unroll 8
  for (int i = 0; i < HDIM / 4; ++i) { float4 v = p[i]; s += v.x*v.x + v.y*v.y + v.z*v.z + v.w*v.w; }
  norms[n] = sqrtf(s);
}

__global__ void k_sim(const float* __restrict__ x, const float* __restrict__ norms,
                      const long long* __restrict__ ei0, const long long* __restrict__ ei1,
                      float* __restrict__ sim) {
  int e = blockIdx.x * blockDim.x + threadIdx.x;
  if (e >= ET) return;
  long long r, c;
  if (e < NE) { r = ei0[e]; c = ei1[e]; } else { r = c = (long long)(e - NE); }
  const float4* pr = (const float4*)(x + (size_t)r * HDIM);
  const float4* pc = (const float4*)(x + (size_t)c * HDIM);
  float s = 0.f;
#pragma unroll 8
  for (int i = 0; i < HDIM / 4; ++i) {
    float4 a = pr[i], b = pc[i];
    s += a.x*b.x + a.y*b.y + a.z*b.z + a.w*b.w;
  }
  float nr = fmaxf(norms[r], COS_EPS), nc = fmaxf(norms[c], COS_EPS);
  sim[e] = s / (nr * nc);
}

__global__ void k_red_init(unsigned* redmax, float* redsum) {
  if (threadIdx.x == 0 && blockIdx.x == 0) { *redmax = f2ord(-FLT_MAX); *redsum = 0.f; }
}

__global__ void k_max(const float* __restrict__ sim, unsigned* __restrict__ redmax) {
  __shared__ unsigned sm[256];
  int tid = threadIdx.x;
  unsigned best = f2ord(-FLT_MAX);
  for (int i = blockIdx.x * 256 + tid; i < ET; i += gridDim.x * 256)
    best = max(best, f2ord(sim[i]));
  sm[tid] = best; __syncthreads();
  for (int s = 128; s > 0; s >>= 1) { if (tid < s) sm[tid] = max(sm[tid], sm[tid + s]); __syncthreads(); }
  if (tid == 0) atomicMax(redmax, sm[0]);
}

__global__ void k_sumexp(const float* __restrict__ sim, const unsigned* __restrict__ redmax,
                         float* __restrict__ redsum) {
  __shared__ float sm[256];
  float m = ord2f(*redmax);
  int tid = threadIdx.x;
  float s = 0.f;
  for (int i = blockIdx.x * 256 + tid; i < ET; i += gridDim.x * 256)
    s += __expf(sim[i] - m);
  sm[tid] = s; __syncthreads();
  for (int t = 128; t > 0; t >>= 1) { if (tid < t) sm[tid] += sm[tid + t]; __syncthreads(); }
  if (tid == 0) unsafeAtomicAdd(redsum, sm[0]);
}

// two edges per 256-thread block, 128 lanes = 128 channels each;
// aggr[col] += softmax_w * lx[row]   (native global_atomic_add_f32)
__global__ __launch_bounds__(256)
void k_scatter(const float* __restrict__ lx, const float* __restrict__ sim,
               const unsigned* __restrict__ redmax, const float* __restrict__ redsum,
               const long long* __restrict__ ei0, const long long* __restrict__ ei1,
               float* __restrict__ aggr) {
  int e = blockIdx.x * 2 + (threadIdx.x >> 7);
  int t = threadIdx.x & 127;
  long long r, c;
  if (e < NE) { r = ei0[e]; c = ei1[e]; } else { r = c = (long long)(e - NE); }
  float w = __expf(sim[e] - ord2f(*redmax)) / (*redsum);
  unsafeAtomicAdd(&aggr[(size_t)c * HDIM + t], w * lx[(size_t)r * HDIM + t]);
}

// x_new = relu(LN(aggr+lx)) (+ sk); write f32 + f16
__global__ __launch_bounds__(128)
void k_combine(const float* __restrict__ aggr, const float* __restrict__ lx,
               const float* __restrict__ sk, int useSkip,
               const float* __restrict__ g, const float* __restrict__ beta,
               float* __restrict__ x, half_t* __restrict__ xh) {
  __shared__ float red[128];
  int n = blockIdx.x, t = threadIdx.x;
  size_t idx = (size_t)n * HDIM + t;
  float a = aggr[idx] + lx[idx];
  red[t] = a; __syncthreads();
  for (int s = 64; s > 0; s >>= 1) { if (t < s) red[t] += red[t + s]; __syncthreads(); }
  float mu = red[0] / HDIM; __syncthreads();
  float d = a - mu;
  red[t] = d * d; __syncthreads();
  for (int s = 64; s > 0; s >>= 1) { if (t < s) red[t] += red[t + s]; __syncthreads(); }
  float rs = rsqrtf(red[0] / HDIM + LN_EPS);
  float v = fmaxf(d * rs * g[t] + beta[t], 0.f);
  if (useSkip) v += sk[idx];
  x[idx] = v;
  xh[idx] = (half_t)v;
}

// edge MLP: concat(user,movie) 256 -> 128 relu -> 64 relu -> 1, + pop bias.
// 32 edges / block; gathers the f16 mirror xh (contiguous 256B rows, b128
// copies), so each weight fragment is loaded once and reused for 2 M-tiles.
__global__ __launch_bounds__(128)
void k_edge_mlp(const half_t* __restrict__ xh,
                const long long* __restrict__ ei0, const long long* __restrict__ ei1,
                const float* __restrict__ pop,
                const half_t* __restrict__ W1, const float* __restrict__ b1,
                const half_t* __restrict__ W2, const float* __restrict__ b2,
                const float* __restrict__ w3, const float* __restrict__ b3,
                const float* __restrict__ popw, const float* __restrict__ popb,
                float* __restrict__ out) {
  __shared__ half_t sA[32 * 256];
  __shared__ half_t sH1[32 * 128];
  __shared__ half_t sH2[32 * 64];
  __shared__ long long sRow[32], sCol[32];

  int e0 = blockIdx.x * 32;
  if (threadIdx.x < 32) {
    long long r = ei0[e0 + threadIdx.x], c = ei1[e0 + threadIdx.x];
    sRow[threadIdx.x] = r; sCol[threadIdx.x] = c;
    __builtin_prefetch(xh + (size_t)r * HDIM, 0, 0);                // global_prefetch_b8
    __builtin_prefetch(xh + (size_t)(N_USER_C + c) * HDIM, 0, 0);
  }
  __syncthreads();
  // stage 32 x 256 f16 tile as 16B chunks: 1024 chunks / 128 threads
  for (int idx = threadIdx.x; idx < 32 * (256 / 8); idx += 128) {
    int r = idx >> 5, c8 = idx & 31;       // 32 chunks (of 8 halves) per row
    const uint4* src = (c8 < 16)
        ? (const uint4*)(xh + (size_t)sRow[r] * HDIM) + c8
        : (const uint4*)(xh + (size_t)(N_USER_C + sCol[r]) * HDIM) + (c8 - 16);
    ((uint4*)sA)[idx] = *src;
  }
  __syncthreads();

  int wave = threadIdx.x >> 5, lane = threadIdx.x & 31;
  // ep1: 256 -> 128, relu. wave owns N-tiles {2w,2w+1}; 2 M-tiles share B frag.
#pragma unroll
  for (int tt = 0; tt < 2; ++tt) {
    int t = wave * 2 + tt;
    v8f acc0 = {}, acc1 = {};
#pragma unroll
    for (int k = 0; k < 8; ++k) {
      v16h bf = load_b16(W1, 256, t * 16, k * 32);
      acc0 = wmma32(load_a16(sA, 256, k * 32), bf, acc0);
      acc1 = wmma32(load_a16(sA + 16 * 256, 256, k * 32), bf, acc1);
    }
    int n = t * 16 + (lane & 15), M0 = (lane >> 4) * 8;
    float bias = b1[n];
#pragma unroll
    for (int r = 0; r < 8; ++r) {
      sH1[(size_t)(M0 + r) * 128 + n]        = (half_t)fmaxf(acc0[r] + bias, 0.f);
      sH1[(size_t)(16 + M0 + r) * 128 + n]   = (half_t)fmaxf(acc1[r] + bias, 0.f);
    }
  }
  __syncthreads();
  // ep2: 128 -> 64, relu. wave owns 1 N-tile; 2 M-tiles share B frag.
  {
    int t = wave;
    v8f acc0 = {}, acc1 = {};
#pragma unroll
    for (int k = 0; k < 4; ++k) {
      v16h bf = load_b16(W2, 128, t * 16, k * 32);
      acc0 = wmma32(load_a16(sH1, 128, k * 32), bf, acc0);
      acc1 = wmma32(load_a16(sH1 + 16 * 128, 128, k * 32), bf, acc1);
    }
    int n = t * 16 + (lane & 15), M0 = (lane >> 4) * 8;
    float bias = b2[n];
#pragma unroll
    for (int r = 0; r < 8; ++r) {
      sH2[(size_t)(M0 + r) * 64 + n]       = (half_t)fmaxf(acc0[r] + bias, 0.f);
      sH2[(size_t)(16 + M0 + r) * 64 + n]  = (half_t)fmaxf(acc1[r] + bias, 0.f);
    }
  }
  __syncthreads();
  // ep3: 64 -> 1, plus popularity bias
  if (threadIdx.x < 32) {
    int e = e0 + threadIdx.x;
    float s = *b3;
#pragma unroll
    for (int k = 0; k < 64; ++k) s += (float)sH2[threadIdx.x * 64 + k] * w3[k];
    s += (*popw) * pop[e] + (*popb);
    out[e] = s;
  }
}

// ---------------- host ------------------------------------------------------
extern "C" void kernel_launch(void* const* d_in, const int* in_sizes, int n_in,
                              void* d_out, int out_size, void* d_ws, size_t ws_size,
                              hipStream_t stream) {
  const float* x_user = (const float*)d_in[0];
  const float* x_movie = (const float*)d_in[1];
  const long long* eidx = (const long long*)d_in[2];
  const long long* ei0 = eidx, * ei1 = eidx + NE;
  const float* pop = (const float*)d_in[3];

  const float* ue_w1 = (const float*)d_in[4];  const float* ue_b1 = (const float*)d_in[5];
  const float* ue_w2 = (const float*)d_in[6];  const float* ue_b2 = (const float*)d_in[7];
  const float* ue_g  = (const float*)d_in[8];  const float* ue_be = (const float*)d_in[9];
  const float* me_w1 = (const float*)d_in[10]; const float* me_b1 = (const float*)d_in[11];
  const float* me_w2 = (const float*)d_in[12]; const float* me_b2 = (const float*)d_in[13];
  const float* me_g  = (const float*)d_in[14]; const float* me_be = (const float*)d_in[15];
  const float* conv_w[3], * conv_b[3], * conv_g[3], * conv_be[3];
  for (int i = 0; i < 3; ++i) {
    conv_w[i]  = (const float*)d_in[16 + 4 * i];
    conv_b[i]  = (const float*)d_in[17 + 4 * i];
    conv_g[i]  = (const float*)d_in[18 + 4 * i];
    conv_be[i] = (const float*)d_in[19 + 4 * i];
  }
  const float* skip_w[2] = { (const float*)d_in[28], (const float*)d_in[30] };
  const float* skip_b[2] = { (const float*)d_in[29], (const float*)d_in[31] };
  const float* ep1_w = (const float*)d_in[32]; const float* ep1_b = (const float*)d_in[33];
  const float* ep2_w = (const float*)d_in[34]; const float* ep2_b = (const float*)d_in[35];
  const float* ep3_w = (const float*)d_in[36]; const float* ep3_b = (const float*)d_in[37];
  const float* pop_w = (const float*)d_in[38]; const float* pop_b = (const float*)d_in[39];
  float* out = (float*)d_out;

  // workspace carve-up
  char* base = (char*)d_ws; size_t off = 0;
  auto carve = [&](size_t bytes) -> void* {
    void* r = base + off; off = (off + bytes + 255) & ~(size_t)255; return r;
  };
  half_t* xh     = (half_t*)carve((size_t)NN * HDIM * 2);
  float*  x      = (float*)carve((size_t)NN * HDIM * 4);
  float*  lx     = (float*)carve((size_t)NN * HDIM * 4);
  float*  aggr   = (float*)carve((size_t)NN * HDIM * 4);
  float*  sk     = (float*)carve((size_t)NN * HDIM * 4);
  float*  norms  = (float*)carve((size_t)NN * 4);
  float*  sim    = (float*)carve((size_t)ET * 4);
  unsigned* redmax = (unsigned*)carve(256);
  float*  redsum = (float*)carve(256);
  half_t* ue_w1h = (half_t*)carve(128 * 32 * 2);
  half_t* me_w1h = (half_t*)carve(128 * 64 * 2);
  half_t* ue_w2h = (half_t*)carve(128 * 128 * 2);
  half_t* me_w2h = (half_t*)carve(128 * 128 * 2);
  half_t* conv_wh[3]; for (int i = 0; i < 3; ++i) conv_wh[i] = (half_t*)carve(128 * 128 * 2);
  half_t* skip_wh[2]; for (int i = 0; i < 2; ++i) skip_wh[i] = (half_t*)carve(128 * 128 * 2);
  half_t* ep1_wh = (half_t*)carve(128 * 256 * 2);
  half_t* ep2_wh = (half_t*)carve(64 * 128 * 2);

  auto cvt = [&](const float* s, half_t* d, int n) {
    k_f32_to_f16<<<(n + 255) / 256, 256, 0, stream>>>(s, d, n);
  };
  cvt(ue_w1, ue_w1h, 128 * 32);   cvt(me_w1, me_w1h, 128 * 64);
  cvt(ue_w2, ue_w2h, 128 * 128);  cvt(me_w2, me_w2h, 128 * 128);
  for (int i = 0; i < 3; ++i) cvt(conv_w[i], conv_wh[i], 128 * 128);
  for (int i = 0; i < 2; ++i) cvt(skip_w[i], skip_wh[i], 128 * 128);
  cvt(ep1_w, ep1_wh, 128 * 256);  cvt(ep2_w, ep2_wh, 64 * 128);

  // encoders
  k_encoder<<<N_USER_C / 16, 128, 0, stream>>>(x_user, 32, N_USER_C, 0,
      ue_w1h, ue_b1, ue_w2h, ue_b2, ue_g, ue_be, x, xh);
  k_encoder<<<N_MOVIE_C / 16, 128, 0, stream>>>(x_movie, 64, N_MOVIE_C, N_USER_C,
      me_w1h, me_b1, me_w2h, me_b2, me_g, me_be, x, xh);

  // conv layers
  for (int i = 0; i < 3; ++i) {
    k_norms<<<(NN + 255) / 256, 256, 0, stream>>>(x, norms);
    k_linear<<<NN / 80, 128, 0, stream>>>(xh, conv_wh[i], conv_b[i], lx);
    if (i > 0)
      k_linear<<<NN / 80, 128, 0, stream>>>(xh, skip_wh[i - 1], skip_b[i - 1], sk);
    k_sim<<<(ET + 255) / 256, 256, 0, stream>>>(x, norms, ei0, ei1, sim);
    k_red_init<<<1, 1, 0, stream>>>(redmax, redsum);
    k_max<<<512, 256, 0, stream>>>(sim, redmax);
    k_sumexp<<<512, 256, 0, stream>>>(sim, redmax, redsum);
    k_zero<<<(NN * HDIM + 255) / 256, 256, 0, stream>>>(aggr, NN * HDIM);
    k_scatter<<<ET / 2, 256, 0, stream>>>(lx, sim, redmax, redsum, ei0, ei1, aggr);
    k_combine<<<NN, 128, 0, stream>>>(aggr, lx, sk, i > 0 ? 1 : 0,
                                      conv_g[i], conv_be[i], x, xh);
  }

  // edge prediction MLP (reads the f16 mirror xh)
  k_edge_mlp<<<NE / 32, 128, 0, stream>>>(xh, ei0, ei1, pop,
      ep1_wh, ep1_b, ep2_wh, ep2_b, ep3_w, ep3_b, pop_w, pop_b, out);
}
